// WorkingMemoryModule_4432406250087
// MI455X (gfx1250) — matrix operand order repaired and checked by reading, not compiled, not run on hardware
//
#include <hip/hip_runtime.h>
#include <hip/hip_bf16.h>

// ---------------------------------------------------------------------------
// Problem constants
// ---------------------------------------------------------------------------
constexpr int Bb = 256;   // batch
constexpr int Ss = 256;   // sequence length
constexpr int Nn = 128;   // memory slots
constexpr int Dd = 256;   // model dim
constexpr int Cc = 512;   // LSTM hidden
constexpr int Hh = 4;     // heads (HD = 64)

typedef __attribute__((ext_vector_type(16))) __bf16 v16bf;
typedef __attribute__((ext_vector_type(8)))  __bf16 v8bf;
typedef __attribute__((ext_vector_type(8)))  float  v8f;

__device__ __forceinline__ float sigmoidf_(float x) { return 1.0f / (1.0f + __expf(-x)); }

// ---------------------------------------------------------------------------
// Generic bf16 WMMA GEMM:  C[z] = A[z] @ W[z]^T (+ bias) (+= if accumulate)
// One wave per 16x16 output tile. grid = (N/16, M/16, batches), block = 32.
// A: (M x K) row-major bf16, W: (N x K) row-major bf16.
// Optional fp32 output C and/or bf16 output Cbf.
// ---------------------------------------------------------------------------
__global__ __launch_bounds__(32) void gemm_bf16_xwT(
    const __bf16* __restrict__ A, int lda, long long sA,
    const __bf16* __restrict__ W, int ldw, long long sW,
    const float* __restrict__ bias,
    float* __restrict__ C, int ldc, long long sC,
    __bf16* __restrict__ Cbf, int ldcbf, long long sCbf,
    int K, int accumulate)
{
    const int lane = threadIdx.x;
    const int row0 = blockIdx.y << 4;
    const int col0 = blockIdx.x << 4;
    const int z    = blockIdx.z;

    const __bf16* Az = A + (size_t)z * sA;
    const __bf16* Wz = W + (size_t)z * sW;

    const int m0 = row0 + ((lane & 16) >> 1);   // base M for C/D fragment
    const int n  = col0 + (lane & 15);          // N column for C/D fragment

    v8f acc = {};
    if (accumulate) {
        const float* Cz0 = C + (size_t)z * sC;
        #pragma unroll
        for (int r = 0; r < 8; ++r) acc[r] = Cz0[(size_t)(m0 + r) * ldc + n];
    }

    // A fragment: lane holds row (lane&15); K chunks [kb,kb+8) and [kb+16,kb+24), kb=(lane>=16)*8
    const __bf16* pa = Az + (size_t)(row0 + (lane & 15)) * lda + ((lane & 16) >> 1);
    // B fragment: lane holds W-row (lane&15); K chunks [kb,kb+8) and [kb+16,kb+24), kb=(lane>=16)*8
    const __bf16* pw = Wz + (size_t)(col0 + (lane & 15)) * ldw + ((lane & 16) >> 1);

    for (int k0 = 0; k0 < K; k0 += 32) {
        v8bf alo = *(const v8bf*)(pa + k0);
        v8bf ahi = *(const v8bf*)(pa + k0 + 16);
        v8bf blo = *(const v8bf*)(pw + k0);
        v8bf bhi = *(const v8bf*)(pw + k0 + 16);
        v16bf af, bfr;
        #pragma unroll
        for (int i = 0; i < 8; ++i) {
            af[i] = alo[i]; af[8 + i] = ahi[i];
            bfr[i] = blo[i]; bfr[8 + i] = bhi[i];
        }
        __builtin_prefetch(pa + k0 + 64, 0, 3);   // global_prefetch_b8
        __builtin_prefetch(pw + k0 + 64, 0, 3);
        acc = __builtin_amdgcn_wmma_f32_16x16x32_bf16(
                  false, af, false, bfr, (short)0, acc, false, false);
    }

    const float bv = bias ? bias[n] : 0.0f;
    float*  Cz  = C   ? C   + (size_t)z * sC   : nullptr;
    __bf16* Cbz = Cbf ? Cbf + (size_t)z * sCbf : nullptr;
    #pragma unroll
    for (int r = 0; r < 8; ++r) {
        float v = acc[r] + bv;
        if (Cz)  Cz [(size_t)(m0 + r) * ldc   + n] = v;
        if (Cbz) Cbz[(size_t)(m0 + r) * ldcbf + n] = (__bf16)v;
    }
}

// ---------------------------------------------------------------------------
// Output GEMM with fused epilogue:
//   out[b,t,:] = x[b,t,:] + sigmoid(rg[b]) * (ctx @ Wout^T + b_out)
// ---------------------------------------------------------------------------
__global__ __launch_bounds__(32) void gemm_out_fused(
    const __bf16* __restrict__ ctxbf, const __bf16* __restrict__ Woutbf,
    const float* __restrict__ bout, const float* __restrict__ rgwg,
    const float* __restrict__ in_seq, float* __restrict__ out_seq, int t)
{
    const int lane = threadIdx.x;
    const int row0 = blockIdx.y << 4;
    const int col0 = blockIdx.x << 4;

    const __bf16* pa = ctxbf  + (size_t)(row0 + (lane & 15)) * Dd + ((lane & 16) >> 1);
    const __bf16* pw = Woutbf + (size_t)(col0 + (lane & 15)) * Dd + ((lane & 16) >> 1);

    v8f acc = {};
    for (int k0 = 0; k0 < Dd; k0 += 32) {
        v8bf alo = *(const v8bf*)(pa + k0);
        v8bf ahi = *(const v8bf*)(pa + k0 + 16);
        v8bf blo = *(const v8bf*)(pw + k0);
        v8bf bhi = *(const v8bf*)(pw + k0 + 16);
        v16bf af, bfr;
        #pragma unroll
        for (int i = 0; i < 8; ++i) {
            af[i] = alo[i]; af[8 + i] = ahi[i];
            bfr[i] = blo[i]; bfr[8 + i] = bhi[i];
        }
        acc = __builtin_amdgcn_wmma_f32_16x16x32_bf16(
                  false, af, false, bfr, (short)0, acc, false, false);
    }

    const int m0 = row0 + ((lane & 16) >> 1);
    const int n  = col0 + (lane & 15);
    #pragma unroll
    for (int r = 0; r < 8; ++r) {
        int m = m0 + r;
        float ro = acc[r] + bout[n];
        float g  = sigmoidf_(rgwg[m * 2]);
        size_t idx = ((size_t)m * Ss + t) * Dd + n;
        out_seq[idx] = in_seq[idx] + g * ro;
    }
}

// ---------------------------------------------------------------------------
// Small helper kernels
// ---------------------------------------------------------------------------
__global__ void f32_to_bf16(const float* __restrict__ s, __bf16* __restrict__ d, int n) {
    int i = blockIdx.x * 256 + threadIdx.x;
    if (i < n) d[i] = (__bf16)s[i];
}

// dst[c*rows + r] = src[r*cols + c]   (Wk -> WkT, with bf16 convert)
__global__ void transpose_to_bf16(const float* __restrict__ s, __bf16* __restrict__ d,
                                  int rows, int cols) {
    int i = blockIdx.x * 256 + threadIdx.x;
    if (i < rows * cols) {
        int r = i / cols, c = i % cols;
        d[(size_t)c * rows + r] = (__bf16)s[i];
    }
}

__global__ void add_bias_vec(const float* a, const float* b, float* o, int n) {
    int i = blockIdx.x * 256 + threadIdx.x;
    if (i < n) o[i] = a[i] + b[i];
}

__global__ void init_mem_kernel(const float* __restrict__ mi, float* __restrict__ mem,
                                __bf16* __restrict__ membf) {
    int i = blockIdx.x * 256 + threadIdx.x;   // Bb*Nn*Dd threads
    float v = mi[i % (Nn * Dd)];
    mem[i] = v; membf[i] = (__bf16)v;
}

__global__ void load_x_bf16(const float* __restrict__ in, __bf16* __restrict__ xbf, int t) {
    int i = blockIdx.x * 256 + threadIdx.x;   // Bb*Dd threads
    int b = i >> 8, j = i & 255;
    xbf[i] = (__bf16)in[((size_t)b * Ss + t) * Dd + j];
}

__global__ void lstm_cell_kernel(const float* __restrict__ gates, float* __restrict__ c,
                                 __bf16* __restrict__ hbf) {
    int i = blockIdx.x * 256 + threadIdx.x;   // Bb*Cc threads
    int b = i >> 9, j = i & 511;
    const float* g = gates + (size_t)b * (4 * Cc);
    float ig = sigmoidf_(g[j]);
    float fg = sigmoidf_(g[Cc + j]);
    float gg = tanhf(g[2 * Cc + j]);
    float og = sigmoidf_(g[3 * Cc + j]);
    float cn = fg * c[i] + ig * gg;
    c[i] = cn;
    hbf[i] = (__bf16)(og * tanhf(cn));
}

// read/write gate logits: rgwg[b*2+0] = h.W_rg + b_rg ; [b*2+1] = h.W_wg + b_wg
__global__ __launch_bounds__(32) void rgwg_kernel(const __bf16* __restrict__ hbf,
                                                  const float* __restrict__ Wrg, const float* brg,
                                                  const float* __restrict__ Wwg, const float* bwg,
                                                  float* __restrict__ rgwg) {
    int b = blockIdx.x, lane = threadIdx.x;
    float s0 = 0.f, s1 = 0.f;
    for (int j = lane; j < Cc; j += 32) {
        float hv = (float)hbf[(size_t)b * Cc + j];
        s0 += hv * Wrg[j];
        s1 += hv * Wwg[j];
    }
    for (int off = 16; off > 0; off >>= 1) {
        s0 += __shfl_down(s0, off, 32);
        s1 += __shfl_down(s1, off, 32);
    }
    if (lane == 0) {
        rgwg[b * 2 + 0] = s0 + brg[0];
        rgwg[b * 2 + 1] = s1 + bwg[0];
    }
}

// U row 4 <- write_key (wp[:, :Dd]) as bf16
__global__ void build_u_row4(const float* __restrict__ wp, __bf16* __restrict__ U) {
    int i = blockIdx.x * 256 + threadIdx.x;   // Bb*Dd
    int b = i >> 8, j = i & 255;
    U[(size_t)b * (16 * Dd) + 4 * Dd + j] = (__bf16)wp[(size_t)b * (2 * Dd) + j];
}

// softmax over n for scores column col (0..3: heads, scale 1/8; 4: write attention)
__global__ void softmax_kernel(const float* __restrict__ scores,
                               float* __restrict__ attn, float* __restrict__ wa) {
    __shared__ float red[Nn];
    int b = blockIdx.x, col = blockIdx.y, n = threadIdx.x;
    float scale = (col < 4) ? 0.125f : 1.0f;   // 1/sqrt(64)
    float v = scores[((size_t)b * Nn + n) * 16 + col] * scale;
    red[n] = v; __syncthreads();
    for (int s = Nn / 2; s > 0; s >>= 1) { if (n < s) red[n] = fmaxf(red[n], red[n + s]); __syncthreads(); }
    float m = red[0]; __syncthreads();
    float e = __expf(v - m);
    red[n] = e; __syncthreads();
    for (int s = Nn / 2; s > 0; s >>= 1) { if (n < s) red[n] += red[n + s]; __syncthreads(); }
    float r = e / red[0];
    if (col < 4) attn[((size_t)b * Hh + col) * Nn + n] = r;
    else         wa[(size_t)b * Nn + n] = r;
}

// mbar[b,h,:] = sum_n attn[b,h,n] * mem[b,n,:]   (bf16 output)
__global__ void mbar_kernel(const float* __restrict__ attn, const float* __restrict__ mem,
                            __bf16* __restrict__ mbarbf) {
    __shared__ float a_s[Nn];
    int b = blockIdx.x, h = blockIdx.y, j = threadIdx.x;   // blockDim = 256
    if (j < Nn) a_s[j] = attn[((size_t)b * Hh + h) * Nn + j];
    __syncthreads();
    const float* mb = mem + (size_t)b * Nn * Dd;
    float acc = 0.f;
    #pragma unroll 4
    for (int n = 0; n < Nn; ++n) acc += a_s[n] * mb[(size_t)n * Dd + j];
    mbarbf[((size_t)b * Hh + h) * Dd + j] = (__bf16)acc;
}

// mem = mem*(1 - wa*sig(erase)) + sig(wg)*(wa*write_val); refresh bf16 shadow
__global__ void mem_update_kernel(float* __restrict__ mem, __bf16* __restrict__ membf,
                                  const float* __restrict__ wa, const float* __restrict__ erase_l,
                                  const float* __restrict__ wp, const float* __restrict__ rgwg) {
    int i = blockIdx.x * 256 + threadIdx.x;   // Bb*Nn*Dd
    int j = i & 255, bn = i >> 8, n = bn & 127, b = bn >> 7;
    float e  = wa[(size_t)b * Nn + n];
    float er = sigmoidf_(erase_l[(size_t)b * Dd + j]);
    float wg = sigmoidf_(rgwg[b * 2 + 1]);
    float wv = wp[(size_t)b * (2 * Dd) + Dd + j];
    float m  = mem[i];
    m = m * (1.f - e * er) + wg * (e * wv);
    mem[i] = m; membf[i] = (__bf16)m;
}

__global__ void copy_mem_out(const float* __restrict__ mem, float* __restrict__ out) {
    int i = blockIdx.x * 256 + threadIdx.x;
    out[i] = mem[i];
}

// ---------------------------------------------------------------------------
// Host driver
// ---------------------------------------------------------------------------
extern "C" void kernel_launch(void* const* d_in, const int* in_sizes, int n_in,
                              void* d_out, int out_size, void* d_ws, size_t ws_size,
                              hipStream_t stream) {
    (void)in_sizes; (void)n_in; (void)out_size; (void)ws_size;

    const float* in_seq   = (const float*)d_in[0];
    const float* mem_init = (const float*)d_in[1];
    const float* W_ih     = (const float*)d_in[2];
    const float* W_hh     = (const float*)d_in[3];
    const float* b_ih     = (const float*)d_in[4];
    const float* b_hh     = (const float*)d_in[5];
    const float* W_read   = (const float*)d_in[6];
    const float* b_read   = (const float*)d_in[7];
    const float* W_write  = (const float*)d_in[8];
    const float* b_write  = (const float*)d_in[9];
    const float* W_erase  = (const float*)d_in[10];
    const float* b_erase  = (const float*)d_in[11];
    const float* W_in     = (const float*)d_in[12];
    const float* b_in     = (const float*)d_in[13];
    const float* W_out    = (const float*)d_in[14];
    const float* b_out    = (const float*)d_in[15];
    const float* W_rg     = (const float*)d_in[16];
    const float* b_rg     = (const float*)d_in[17];
    const float* W_wg     = (const float*)d_in[18];
    const float* b_wg     = (const float*)d_in[19];

    float* out_seq = (float*)d_out;
    float* out_mem = (float*)d_out + (size_t)Bb * Ss * Dd;

    // --- workspace allocator ---
    char* wptr = (char*)d_ws;
    auto alloc = [&](size_t bytes) -> void* {
        void* p = (void*)wptr;
        wptr += (bytes + 255) & ~(size_t)255;
        return p;
    };

    __bf16* Wih_bf   = (__bf16*)alloc((size_t)4 * Cc * Dd * 2);
    __bf16* Whh_bf   = (__bf16*)alloc((size_t)4 * Cc * Cc * 2);
    __bf16* Wread_bf = (__bf16*)alloc((size_t)Dd * Cc * 2);
    __bf16* Wwrite_bf= (__bf16*)alloc((size_t)2 * Dd * Cc * 2);
    __bf16* Werase_bf= (__bf16*)alloc((size_t)Dd * Cc * 2);
    __bf16* Wq_bf    = (__bf16*)alloc((size_t)Dd * Dd * 2);
    __bf16* WkT_bf   = (__bf16*)alloc((size_t)Dd * Dd * 2);
    __bf16* Wv_bf    = (__bf16*)alloc((size_t)Dd * Dd * 2);
    __bf16* Wout_bf  = (__bf16*)alloc((size_t)Dd * Dd * 2);
    float*  bias_sum = (float*) alloc((size_t)4 * Cc * 4);

    __bf16* hbf   = (__bf16*)alloc((size_t)Bb * Cc * 2);
    float*  cstate= (float*) alloc((size_t)Bb * Cc * 4);
    float*  gates = (float*) alloc((size_t)Bb * 4 * Cc * 4);
    __bf16* xbf   = (__bf16*)alloc((size_t)Bb * Dd * 2);
    __bf16* rkbf  = (__bf16*)alloc((size_t)Bb * Dd * 2);
    __bf16* qbf   = (__bf16*)alloc((size_t)Bb * Dd * 2);
    float*  wp    = (float*) alloc((size_t)Bb * 2 * Dd * 4);
    float*  erase = (float*) alloc((size_t)Bb * Dd * 4);
    float*  rgwg  = (float*) alloc((size_t)Bb * 2 * 4);
    __bf16* Ubf   = (__bf16*)alloc((size_t)Bb * 16 * Dd * 2);
    float*  scores= (float*) alloc((size_t)Bb * Nn * 16 * 4);
    float*  attn  = (float*) alloc((size_t)Bb * Hh * Nn * 4);
    float*  wa    = (float*) alloc((size_t)Bb * Nn * 4);
    __bf16* mbarbf= (__bf16*)alloc((size_t)Bb * Hh * Dd * 2);
    __bf16* ctxbf = (__bf16*)alloc((size_t)Bb * Dd * 2);
    float*  mem   = (float*) alloc((size_t)Bb * Nn * Dd * 4);
    __bf16* membf = (__bf16*)alloc((size_t)Bb * Nn * Dd * 2);

    const dim3 b256(256), b32(32), b128(128);
    auto blks = [](int n) { return dim3((n + 255) / 256); };

    // --- one-time init ---
    f32_to_bf16<<<blks(4*Cc*Dd), b256, 0, stream>>>(W_ih, Wih_bf, 4*Cc*Dd);
    f32_to_bf16<<<blks(4*Cc*Cc), b256, 0, stream>>>(W_hh, Whh_bf, 4*Cc*Cc);
    f32_to_bf16<<<blks(Dd*Cc),   b256, 0, stream>>>(W_read, Wread_bf, Dd*Cc);
    f32_to_bf16<<<blks(2*Dd*Cc), b256, 0, stream>>>(W_write, Wwrite_bf, 2*Dd*Cc);
    f32_to_bf16<<<blks(Dd*Cc),   b256, 0, stream>>>(W_erase, Werase_bf, Dd*Cc);
    f32_to_bf16<<<blks(Dd*Dd),   b256, 0, stream>>>(W_in, Wq_bf, Dd*Dd);                 // Wq
    transpose_to_bf16<<<blks(Dd*Dd), b256, 0, stream>>>(W_in + (size_t)Dd*Dd, WkT_bf, Dd, Dd); // Wk^T
    f32_to_bf16<<<blks(Dd*Dd),   b256, 0, stream>>>(W_in + (size_t)2*Dd*Dd, Wv_bf, Dd*Dd); // Wv
    f32_to_bf16<<<blks(Dd*Dd),   b256, 0, stream>>>(W_out, Wout_bf, Dd*Dd);
    add_bias_vec<<<blks(4*Cc), b256, 0, stream>>>(b_ih, b_hh, bias_sum, 4*Cc);

    hipMemsetAsync(hbf,    0, (size_t)Bb * Cc * 2, stream);
    hipMemsetAsync(cstate, 0, (size_t)Bb * Cc * 4, stream);
    hipMemsetAsync(Ubf,    0, (size_t)Bb * 16 * Dd * 2, stream);   // rows 5..15 stay zero
    init_mem_kernel<<<blks(Bb*Nn*Dd), b256, 0, stream>>>(mem_init, mem, membf);

    // --- timestep loop ---
    for (int t = 0; t < Ss; ++t) {
        load_x_bf16<<<blks(Bb*Dd), b256, 0, stream>>>(in_seq, xbf, t);

        // gates = x@Wih^T + bias_sum ; gates += h@Whh^T
        gemm_bf16_xwT<<<dim3(4*Cc/16, Bb/16, 1), b32, 0, stream>>>(
            xbf, Dd, 0, Wih_bf, Dd, 0, bias_sum, gates, 4*Cc, 0,
            nullptr, 0, 0, Dd, 0);
        gemm_bf16_xwT<<<dim3(4*Cc/16, Bb/16, 1), b32, 0, stream>>>(
            hbf, Cc, 0, Whh_bf, Cc, 0, nullptr, gates, 4*Cc, 0,
            nullptr, 0, 0, Cc, 1);
        lstm_cell_kernel<<<blks(Bb*Cc), b256, 0, stream>>>(gates, cstate, hbf);

        // rk (bf16 only), wp (fp32), erase (fp32)
        gemm_bf16_xwT<<<dim3(Dd/16, Bb/16, 1), b32, 0, stream>>>(
            hbf, Cc, 0, Wread_bf, Cc, 0, b_read, nullptr, 0, 0,
            rkbf, Dd, 0, Cc, 0);
        gemm_bf16_xwT<<<dim3(2*Dd/16, Bb/16, 1), b32, 0, stream>>>(
            hbf, Cc, 0, Wwrite_bf, Cc, 0, b_write, wp, 2*Dd, 0,
            nullptr, 0, 0, Cc, 0);
        gemm_bf16_xwT<<<dim3(Dd/16, Bb/16, 1), b32, 0, stream>>>(
            hbf, Cc, 0, Werase_bf, Cc, 0, b_erase, erase, Dd, 0,
            nullptr, 0, 0, Cc, 0);
        rgwg_kernel<<<dim3(Bb), b32, 0, stream>>>(hbf, W_rg, b_rg, W_wg, b_wg, rgwg);

        // q = rk@Wq^T + bq  (bf16)
        gemm_bf16_xwT<<<dim3(Dd/16, Bb/16, 1), b32, 0, stream>>>(
            rkbf, Dd, 0, Wq_bf, Dd, 0, b_in, nullptr, 0, 0,
            qbf, Dd, 0, Dd, 0);

        // tk[b,h,:] = q_head @ Wk_head  -> rows 0..3 of U  (bk dropped: softmax-invariant)
        for (int h = 0; h < Hh; ++h) {
            gemm_bf16_xwT<<<dim3(Dd/16, Bb/16, 1), b32, 0, stream>>>(
                qbf + h * 64, Dd, 0, WkT_bf + h * 64, Dd, 0, nullptr,
                nullptr, 0, 0, Ubf + (size_t)h * Dd, 16 * Dd, 0, 64, 0);
        }
        build_u_row4<<<blks(Bb*Dd), b256, 0, stream>>>(wp, Ubf);   // row 4 = write key

        // scores[b] (Nn x 16) = mem_b @ U_b^T  (batched over b)
        gemm_bf16_xwT<<<dim3(1, Nn/16, Bb), b32, 0, stream>>>(
            membf, Dd, (long long)Nn * Dd, Ubf, Dd, (long long)16 * Dd, nullptr,
            scores, 16, (long long)Nn * 16, nullptr, 0, 0, Dd, 0);

        softmax_kernel<<<dim3(Bb, 5), b128, 0, stream>>>(scores, attn, wa);
        mbar_kernel<<<dim3(Bb, Hh), b256, 0, stream>>>(attn, mem, mbarbf);
        mem_update_kernel<<<blks(Bb*Nn*Dd), b256, 0, stream>>>(mem, membf, wa, erase, wp, rgwg);

        // ctx[:, head h] = mbar[:,h,:] @ Wv_head^T + bv_head  (bv via attn-sum==1)
        for (int h = 0; h < Hh; ++h) {
            gemm_bf16_xwT<<<dim3(64/16, Bb/16, 1), b32, 0, stream>>>(
                mbarbf + (size_t)h * Dd, Hh * Dd, 0,
                Wv_bf + (size_t)h * 64 * Dd, Dd, 0,
                b_in + 2 * Dd + h * 64,
                nullptr, 0, 0, ctxbf + h * 64, Dd, 0, Dd, 0);
        }

        // out[b,t,:] = x + sigmoid(rg) * (ctx @ Wout^T + b_out)
        gemm_out_fused<<<dim3(Dd/16, Bb/16), b32, 0, stream>>>(
            ctxbf, Wout_bf, b_out, rgwg, in_seq, out_seq, t);
    }

    copy_mem_out<<<blks(Bb*Nn*Dd), b256, 0, stream>>>(mem, out_mem);
}
